// PeriodicGraphNeuralNetwork_7275674599803
// MI455X (gfx1250) — matrix operand order repaired and checked by reading, not compiled, not run on hardware
//
#include <hip/hip_runtime.h>

// ---------------- problem constants (match reference) ----------------
#define NAT   2048
#define NDIM  128          // node_dim
#define EDIM  64           // edge_dim / num_rbf
#define HDIM  256          // hidden_dim
#define NL    4            // layers
#define KNB   24           // max neighbors
#define EDG   (NAT*KNB)    // 49152 edges
#define CUTF  8.0f
#define PI_F  3.14159265358979f
#define NTW   4            // 16-col n-tiles per wave (wave computes 16x64 strip)

typedef __attribute__((ext_vector_type(16))) __bf16 v16bf;
typedef __attribute__((ext_vector_type(8)))  float  v8f;

__device__ __forceinline__ unsigned short f2bf(float f) {
  unsigned u = __float_as_uint(f);
  u += 0x7fffu + ((u >> 16) & 1u);          // round-to-nearest-even
  return (unsigned short)(u >> 16);
}
__device__ __forceinline__ float bf2f(unsigned short h) {
  return __uint_as_float(((unsigned)h) << 16);
}
__device__ __forceinline__ float siluf(float x) { return x / (1.f + __expf(-x)); }

// ---------------- lattice + cartesian coords ----------------
// latsv: [0..8] = lattice (row-major 3x3), [9..89] = 27 shift vectors (s*3+d)
__global__ void pg_lattice(const float* __restrict__ lengths, const float* __restrict__ angles,
                           float* __restrict__ latsv) {
  if (threadIdx.x != 0 || blockIdx.x != 0) return;
  float a = lengths[0], b = lengths[1], c = lengths[2];
  float deg = PI_F / 180.f;
  float al = angles[0] * deg, be = angles[1] * deg, ga = angles[2] * deg;
  float lx = a;
  float xy = b * __cosf(ga);
  float xz = c * __cosf(be);
  float ly = b * __sinf(ga);
  float yz = (b * c * __cosf(al) - xy * xz) / ly;
  float lz = sqrtf(c * c - xz * xz - yz * yz);
  float L[9] = { lx, 0.f, 0.f,  xy, ly, 0.f,  xz, yz, lz };
  for (int i = 0; i < 9; ++i) latsv[i] = L[i];
  for (int ii = -1; ii <= 1; ++ii)
    for (int jj = -1; jj <= 1; ++jj)
      for (int kk = -1; kk <= 1; ++kk) {
        int s = (ii + 1) * 9 + (jj + 1) * 3 + (kk + 1);
        for (int d = 0; d < 3; ++d)
          latsv[9 + s * 3 + d] = ii * L[0 * 3 + d] + jj * L[1 * 3 + d] + kk * L[2 * 3 + d];
      }
}

__global__ void pg_cart(const float* __restrict__ frac, const float* __restrict__ latsv,
                        float* __restrict__ cart) {
  int i = blockIdx.x * blockDim.x + threadIdx.x;
  if (i >= NAT) return;
  float f0 = frac[i * 3], f1 = frac[i * 3 + 1], f2 = frac[i * 3 + 2];
  for (int d = 0; d < 3; ++d)
    cart[i * 3 + d] = f0 * latsv[0 * 3 + d] + f1 * latsv[1 * 3 + d] + f2 * latsv[2 * 3 + d];
}

// ---------------- periodic top-K neighbor search ----------------
// one 256-thread block per atom i; per-thread register top-24, then 24 LDS argmin rounds
__global__ void pg_neighbors(const float* __restrict__ cart, const float* __restrict__ latsv,
                             int* __restrict__ nbr, float* __restrict__ dist,
                             float* __restrict__ mask) {
  __shared__ float sdist[256 * KNB];
  __shared__ int   sidx [256 * KNB];
  __shared__ float rd[256];
  __shared__ int   ri[256];
  __shared__ float ssv[81];
  int i = blockIdx.x, t = threadIdx.x;
  if (t < 81) ssv[t] = latsv[9 + t];
  __syncthreads();
  float xi = cart[i * 3], yi = cart[i * 3 + 1], zi = cart[i * 3 + 2];
  float bd[KNB]; int bj[KNB];
#pragma unroll
  for (int k = 0; k < KNB; ++k) { bd[k] = __builtin_inff(); bj[k] = 0; }
  for (int s = 0; s < 27; ++s) {
    float sx = ssv[s * 3], sy = ssv[s * 3 + 1], sz = ssv[s * 3 + 2];
    for (int j = t; j < NAT; j += 256) {
      float dx = cart[j * 3]     + sx - xi;
      float dy = cart[j * 3 + 1] + sy - yi;
      float dz = cart[j * 3 + 2] + sz - zi;
      float d2 = dx * dx + dy * dy + dz * dz;
      float d  = sqrtf(fmaxf(d2, 1e-12f));
      if (d < CUTF && d > 0.01f && d < bd[KNB - 1]) {
        bd[KNB - 1] = d; bj[KNB - 1] = s * NAT + j;
#pragma unroll
        for (int k = KNB - 1; k > 0; --k) {
          if (bd[k] < bd[k - 1]) {
            float td = bd[k]; bd[k] = bd[k - 1]; bd[k - 1] = td;
            int   tj = bj[k]; bj[k] = bj[k - 1]; bj[k - 1] = tj;
          }
        }
      }
    }
  }
#pragma unroll
  for (int k = 0; k < KNB; ++k) { sdist[t * KNB + k] = bd[k]; sidx[t * KNB + k] = bj[k]; }
  __syncthreads();
  for (int kk = 0; kk < KNB; ++kk) {
    float md = __builtin_inff(); int mi = -1;
    for (int q = t; q < 256 * KNB; q += 256) {
      float d = sdist[q];
      if (d < md) { md = d; mi = q; }
    }
    rd[t] = md; ri[t] = mi;
    __syncthreads();
    for (int stp = 128; stp > 0; stp >>= 1) {
      if (t < stp) {
        if (rd[t + stp] < rd[t]) { rd[t] = rd[t + stp]; ri[t] = ri[t + stp]; }
      }
      __syncthreads();
    }
    if (t == 0) {
      int e = i * KNB + kk;
      float d = rd[0]; int q = ri[0];
      if (q >= 0 && d < 1e30f) {
        nbr[e] = sidx[q] % NAT; dist[e] = d; mask[e] = 1.f;
        sdist[q] = __builtin_inff();
      } else { nbr[e] = 0; dist[e] = 0.f; mask[e] = 0.f; }
    }
    __syncthreads();
  }
}

// ---------------- RBF edge features (bf16) ----------------
__global__ void pg_rbf(const float* __restrict__ dist, unsigned short* __restrict__ rbf) {
  int idx = blockIdx.x * blockDim.x + threadIdx.x;
  if (idx >= EDG * EDIM) return;
  int e = idx / EDIM, c = idx % EDIM;
  float d = dist[e];
  float center = CUTF * (float)c / (float)(EDIM - 1);
  float g  = __expf(-(d - center) * (d - center) * 32.f);  // 1/(2*(8/64)^2)=32
  float cv = 0.5f * (__cosf(d * PI_F / CUTF) + 1.f) * (d < CUTF ? 1.f : 0.f);
  rbf[idx] = f2bf(g * cv);
}

// ---------------- f32 -> bf16 bulk convert ----------------
__global__ void pg_f2b(const float* __restrict__ src, unsigned short* __restrict__ dst, int n) {
  int i = blockIdx.x * blockDim.x + threadIdx.x;
  if (i < n) dst[i] = f2bf(src[i]);
}

__global__ void pg_zero(float* __restrict__ p, int n) {
  int i = blockIdx.x * blockDim.x + threadIdx.x;
  if (i < n) p[i] = 0.f;
}

// ---------------- weight repack into WMMA B-fragment layout ----------------
// B (32x16, bf16): VGPR q, K = 8*half + 2*(q&3) + 16*(q>>2) +{0,1} (half = lane>>4); N = lane&15
// packed dword index = ((nt*Kc + c)*32 + lane)*8 + q
__global__ void pg_pack(const float* __restrict__ W, unsigned* __restrict__ out, int K, int N) {
  int t = blockIdx.x * blockDim.x + threadIdx.x;
  if (t >= (K * N) / 2) return;
  int Kc = K >> 5;
  int q = t & 7, lane = (t >> 3) & 31, rest = t >> 8;
  int c = rest % Kc, nt = rest / Kc;
  int half = lane >> 4;
  int n = nt * 16 + (lane & 15);
  int k0 = c * 32 + ((q & 3) << 1) + (q >> 2) * 16 + half * 8;
  unsigned lo = f2bf(W[(size_t)k0 * N + n]);
  unsigned hi = f2bf(W[(size_t)(k0 + 1) * N + n]);
  out[t] = lo | (hi << 16);
}

// ---------------- generic WMMA bf16 GEMM: C = act(A@B + bias) ----------------
// A: [M,K] bf16 row-major; Bp: packed weights; one wave -> 16x64 output strip
// (one A fragment feeds NTW=4 WMMAs per K-chunk -> 4x less A traffic)
__global__ void pg_gemm(const unsigned short* __restrict__ A, const unsigned* __restrict__ Bp,
                        const float* __restrict__ bias, unsigned short* __restrict__ Cb,
                        float* __restrict__ Cf, int M, int K, int N, int act, int resid) {
  int wave = (int)((blockIdx.x * blockDim.x + threadIdx.x) >> 5);
  int lane = threadIdx.x & 31;
  int Mt = M >> 4, Ns = N >> 6;               // 64-column strips
  if (wave >= Mt * Ns) return;                // whole-wave uniform exit: EXEC all-ones for WMMA
  int mt = wave / Ns, ns = wave % Ns;
  int half = lane >> 4, l15 = lane & 15;
  int Kc = K >> 5;
  const unsigned short* arow  = A  + (size_t)(mt * 16 + l15) * K + half * 8;
  const unsigned*       bbase = Bp + ((size_t)(ns * NTW) * Kc * 32 + lane) * 8;

  v8f acc[NTW];
#pragma unroll
  for (int t = 0; t < NTW; ++t) {
    float bv = bias ? bias[ns * 64 + t * 16 + l15] : 0.f;
#pragma unroll
    for (int r = 0; r < 8; ++r) acc[t][r] = bv;
  }

  for (int c = 0; c < Kc; ++c) {
    union { v16bf v; uint4 q[2]; } fa;
    fa.q[0] = *(const uint4*)(arow + c * 32);        // K = half*8 .. +7
    fa.q[1] = *(const uint4*)(arow + c * 32 + 16);   // K = 16+half*8 .. +7
#pragma unroll
    for (int t = 0; t < NTW; ++t) {
      union { v16bf v; uint4 q[2]; } fb;
      const unsigned* bt = bbase + ((size_t)t * Kc + c) * 256;
      fb.q[0] = *(const uint4*)(bt);
      fb.q[1] = *(const uint4*)(bt + 4);
      acc[t] = __builtin_amdgcn_wmma_f32_16x16x32_bf16(false, fa.v, false, fb.v,
                                                       (short)0, acc[t], false, false);
    }
  }

#pragma unroll
  for (int t = 0; t < NTW; ++t) {
    int n = ns * 64 + t * 16 + l15;
#pragma unroll
    for (int r = 0; r < 8; ++r) {
      int m = mt * 16 + r + half * 8;
      float v = acc[t][r];
      if (act) v = siluf(v);
      size_t o = (size_t)m * N + n;
      float outv = v;
      if (Cf) {
        if (resid) outv = Cf[o] + v;
        Cf[o] = outv;
      }
      if (Cb) Cb[o] = f2bf(outv);
    }
  }
}

// ---------------- node feature init: nf += emb[atom_type]; nfb = bf(nf) ----------------
__global__ void pg_add_emb(float* __restrict__ nf, unsigned short* __restrict__ nfb,
                           const int* __restrict__ at, const float* __restrict__ emb) {
  int idx = blockIdx.x * blockDim.x + threadIdx.x;
  if (idx >= NAT * NDIM) return;
  int i = idx >> 7, c = idx & 127;
  float v = nf[idx] + emb[at[i] * NDIM + c];
  nf[idx] = v;
  nfb[idx] = f2bf(v);
}

// ---------------- edge feature gather: ef = [nf[src] | nf[dst] | rbf] ----------------
__global__ void pg_gather_ef(const unsigned short* __restrict__ nfb, const int* __restrict__ nbr,
                             const unsigned short* __restrict__ rbf, unsigned short* __restrict__ ef) {
  int idx = blockIdx.x * blockDim.x + threadIdx.x;
  if (idx >= EDG * (2 * NDIM + EDIM)) return;
  int e = idx / (2 * NDIM + EDIM), c = idx % (2 * NDIM + EDIM);
  unsigned short v;
  if (c < NDIM)            v = nfb[(e / KNB) * NDIM + c];
  else if (c < 2 * NDIM)   v = nfb[nbr[e] * NDIM + (c - NDIM)];
  else                     v = rbf[e * EDIM + (c - 2 * NDIM)];
  ef[idx] = v;
}

// ---------------- aggr[dst] += mask * msg ----------------
__global__ void pg_aggr(const unsigned short* __restrict__ msg, const int* __restrict__ nbr,
                        const float* __restrict__ mask, float* __restrict__ aggr) {
  int idx = blockIdx.x * blockDim.x + threadIdx.x;
  if (idx >= EDG * HDIM) return;
  int e = idx >> 8, h = idx & (HDIM - 1);
  float mv = mask[e];
  if (mv != 0.f) atomicAdd(&aggr[(size_t)nbr[e] * HDIM + h], bf2f(msg[idx]));
}

// ---------------- X = [nf | aggr] bf16 ----------------
__global__ void pg_concatX(const unsigned short* __restrict__ nfb, const float* __restrict__ aggr,
                           unsigned short* __restrict__ X) {
  int idx = blockIdx.x * blockDim.x + threadIdx.x;
  if (idx >= NAT * (NDIM + HDIM)) return;
  int i = idx / (NDIM + HDIM), c = idx % (NDIM + HDIM);
  X[idx] = (c < NDIM) ? nfb[i * NDIM + c] : f2bf(aggr[(size_t)i * HDIM + (c - NDIM)]);
}

// ---------------- w[e] = w1[e,:] . cW2 + cb2 ----------------
__global__ void pg_wdot(const unsigned short* __restrict__ w1, const float* __restrict__ cW2,
                        const float* __restrict__ cb2, float* __restrict__ wv) {
  int e = blockIdx.x * blockDim.x + threadIdx.x;
  if (e >= EDG) return;
  float s = cb2[0];
  const unsigned short* row = w1 + (size_t)e * (HDIM / 2);
  for (int h = 0; h < HDIM / 2; ++h) s += bf2f(row[h]) * cW2[h];
  wv[e] = s;
}

// ---------------- coord scatter: delta[dst] += w * normalize(c_src - c_dst) * mask ----------------
__global__ void pg_coord(const float* __restrict__ coords, const int* __restrict__ nbr,
                         const float* __restrict__ wv, const float* __restrict__ mask,
                         float* __restrict__ delta) {
  int e = blockIdx.x * blockDim.x + threadIdx.x;
  if (e >= EDG) return;
  float mv = mask[e];
  if (mv == 0.f) return;
  int s = e / KNB, d = nbr[e];
  float dx = coords[s * 3]     - coords[d * 3];
  float dy = coords[s * 3 + 1] - coords[d * 3 + 1];
  float dz = coords[s * 3 + 2] - coords[d * 3 + 2];
  float nrm = sqrtf(dx * dx + dy * dy + dz * dz) + 1e-8f;
  float w = wv[e] * mv / nrm;
  atomicAdd(&delta[d * 3],     w * dx);
  atomicAdd(&delta[d * 3 + 1], w * dy);
  atomicAdd(&delta[d * 3 + 2], w * dz);
}

__global__ void pg_apply(float* __restrict__ coords, const float* __restrict__ delta) {
  int i = blockIdx.x * blockDim.x + threadIdx.x;
  if (i < NAT * 3) coords[i] += delta[i];
}

// ---------------- final output: [nf (N*128) | coords (N*3)] ----------------
__global__ void pg_out(const float* __restrict__ nf, const float* __restrict__ coords,
                       float* __restrict__ out) {
  int i = blockIdx.x * blockDim.x + threadIdx.x;
  if (i < NAT * NDIM) out[i] = nf[i];
  else if (i < NAT * NDIM + NAT * 3) out[i] = coords[i - NAT * NDIM];
}

// =======================================================================
extern "C" void kernel_launch(void* const* d_in, const int* in_sizes, int n_in,
                              void* d_out, int out_size, void* d_ws, size_t ws_size,
                              hipStream_t stream) {
  (void)in_sizes; (void)n_in; (void)out_size; (void)ws_size;
  const int*   at      = (const int*)  d_in[0];
  const float* frac    = (const float*)d_in[1];
  const float* lengths = (const float*)d_in[2];
  const float* angles  = (const float*)d_in[3];
  const float* ts      = (const float*)d_in[4];
  const float* emb     = (const float*)d_in[5];
  const float* tW      = (const float*)d_in[6];
  const float* tb      = (const float*)d_in[7];
  const float* eW1     = (const float*)d_in[8];
  const float* eb1     = (const float*)d_in[9];
  const float* eW2     = (const float*)d_in[10];
  const float* eb2     = (const float*)d_in[11];
  const float* nW1     = (const float*)d_in[12];
  const float* nb1     = (const float*)d_in[13];
  const float* nW2     = (const float*)d_in[14];
  const float* nb2     = (const float*)d_in[15];
  const float* cW1     = (const float*)d_in[16];
  const float* cb1     = (const float*)d_in[17];
  const float* cW2     = (const float*)d_in[18];
  const float* cb2     = (const float*)d_in[19];

  // ---- workspace carve-out (256B aligned) ----
  char* wp = (char*)d_ws;
  auto alloc = [&](size_t bytes) -> char* {
    char* p = wp; wp += (bytes + 255) & ~(size_t)255; return p;
  };
  float* latsv  = (float*)alloc(512);
  float* coords = (float*)alloc((size_t)NAT * 3 * 4);      // cart, updated in place
  int*   nbr    = (int*)  alloc((size_t)EDG * 4);
  float* dist   = (float*)alloc((size_t)EDG * 4);
  float* mask   = (float*)alloc((size_t)EDG * 4);
  unsigned short* rbf  = (unsigned short*)alloc((size_t)EDG * EDIM * 2);
  unsigned short* tsb  = (unsigned short*)alloc((size_t)NAT * NDIM * 2);
  float* nf            = (float*)alloc((size_t)NAT * NDIM * 4);
  unsigned short* nfb  = (unsigned short*)alloc((size_t)NAT * NDIM * 2);
  float* aggr          = (float*)alloc((size_t)NAT * HDIM * 4);
  unsigned short* Xb   = (unsigned short*)alloc((size_t)NAT * (NDIM + HDIM) * 2);
  unsigned short* Tb   = (unsigned short*)alloc((size_t)NAT * HDIM * 2);
  float* wv            = (float*)alloc((size_t)EDG * 4);
  float* delta         = (float*)alloc((size_t)NAT * 3 * 4);
  unsigned short* B0   = (unsigned short*)alloc((size_t)EDG * (2 * NDIM + EDIM) * 2); // ef / msg
  unsigned short* B1   = (unsigned short*)alloc((size_t)EDG * HDIM * 2);              // H1 / w1
  unsigned* tWp  = (unsigned*)alloc((size_t)NDIM * NDIM * 2);
  unsigned* eW1p = (unsigned*)alloc((size_t)NL * (2 * NDIM + EDIM) * HDIM * 2);
  unsigned* eW2p = (unsigned*)alloc((size_t)NL * HDIM * HDIM * 2);
  unsigned* nW1p = (unsigned*)alloc((size_t)NL * (NDIM + HDIM) * HDIM * 2);
  unsigned* nW2p = (unsigned*)alloc((size_t)NL * HDIM * NDIM * 2);
  unsigned* cW1p = (unsigned*)alloc((size_t)NL * HDIM * (HDIM / 2) * 2);

  auto cdiv = [](long a, long b) { return (unsigned)((a + b - 1) / b); };
  auto gemm = [&](const unsigned short* A, const unsigned* Bp, const float* bias,
                  unsigned short* Cb, float* Cf, int M, int K, int N, int act, int resid) {
    long waves = (long)(M / 16) * (N / 64);          // 16x64 strip per wave
    pg_gemm<<<cdiv(waves, 8), 256, 0, stream>>>(A, Bp, bias, Cb, Cf, M, K, N, act, resid);
  };
  auto pack = [&](const float* W, unsigned* out, int K, int N) {
    pg_pack<<<cdiv((long)K * N / 2, 256), 256, 0, stream>>>(W, out, K, N);
  };

  // ---- graph construction ----
  pg_lattice<<<1, 1, 0, stream>>>(lengths, angles, latsv);
  pg_cart<<<cdiv(NAT, 256), 256, 0, stream>>>(frac, latsv, coords);
  pg_neighbors<<<NAT, 256, 0, stream>>>(coords, latsv, nbr, dist, mask);
  pg_rbf<<<cdiv((long)EDG * EDIM, 256), 256, 0, stream>>>(dist, rbf);

  // ---- weight repack (WMMA B layout, bf16) ----
  pack(tW, tWp, NDIM, NDIM);
  for (int l = 0; l < NL; ++l) {
    pack(eW1 + (size_t)l * (2 * NDIM + EDIM) * HDIM, eW1p + (size_t)l * (2 * NDIM + EDIM) * HDIM / 2, 2 * NDIM + EDIM, HDIM);
    pack(eW2 + (size_t)l * HDIM * HDIM,              eW2p + (size_t)l * HDIM * HDIM / 2,              HDIM, HDIM);
    pack(nW1 + (size_t)l * (NDIM + HDIM) * HDIM,     nW1p + (size_t)l * (NDIM + HDIM) * HDIM / 2,     NDIM + HDIM, HDIM);
    pack(nW2 + (size_t)l * HDIM * NDIM,              nW2p + (size_t)l * HDIM * NDIM / 2,              HDIM, NDIM);
    pack(cW1 + (size_t)l * HDIM * (HDIM / 2),        cW1p + (size_t)l * HDIM * (HDIM / 2) / 2,        HDIM, HDIM / 2);
  }

  // ---- node init: nf = timesteps @ tW + tb + emb[atom_types] ----
  pg_f2b<<<cdiv((long)NAT * NDIM, 256), 256, 0, stream>>>(ts, tsb, NAT * NDIM);
  gemm(tsb, tWp, tb, nullptr, nf, NAT, NDIM, NDIM, 0, 0);
  pg_add_emb<<<cdiv((long)NAT * NDIM, 256), 256, 0, stream>>>(nf, nfb, at, emb);

  // ---- EGNN layers ----
  for (int l = 0; l < NL; ++l) {
    const float* eb1l = eb1 + l * HDIM;
    const float* eb2l = eb2 + l * HDIM;
    const float* nb1l = nb1 + l * HDIM;
    const float* nb2l = nb2 + l * NDIM;
    const float* cb1l = cb1 + l * (HDIM / 2);
    const unsigned* eW1pl = eW1p + (size_t)l * (2 * NDIM + EDIM) * HDIM / 2;
    const unsigned* eW2pl = eW2p + (size_t)l * HDIM * HDIM / 2;
    const unsigned* nW1pl = nW1p + (size_t)l * (NDIM + HDIM) * HDIM / 2;
    const unsigned* nW2pl = nW2p + (size_t)l * HDIM * NDIM / 2;
    const unsigned* cW1pl = cW1p + (size_t)l * HDIM * (HDIM / 2) / 2;

    // edge MLP: ef -> H1 -> msg   (msg reuses the ef buffer)
    pg_gather_ef<<<cdiv((long)EDG * (2 * NDIM + EDIM), 256), 256, 0, stream>>>(nfb, nbr, rbf, B0);
    gemm(B0, eW1pl, eb1l, B1, nullptr, EDG, 2 * NDIM + EDIM, HDIM, 1, 0);
    gemm(B1, eW2pl, eb2l, B0, nullptr, EDG, HDIM, HDIM, 1, 0);           // msg = B0
    unsigned short* msg = B0;

    // aggregate messages to dst nodes
    pg_zero<<<cdiv((long)NAT * HDIM, 256), 256, 0, stream>>>(aggr, NAT * HDIM);
    pg_aggr<<<cdiv((long)EDG * HDIM, 256), 256, 0, stream>>>(msg, nbr, mask, aggr);

    // node MLP with residual: nf += silu([nf|aggr] @ nW1 + nb1) @ nW2 + nb2
    pg_concatX<<<cdiv((long)NAT * (NDIM + HDIM), 256), 256, 0, stream>>>(nfb, aggr, Xb);
    gemm(Xb, nW1pl, nb1l, Tb, nullptr, NAT, NDIM + HDIM, HDIM, 1, 0);
    gemm(Tb, nW2pl, nb2l, nfb, nf, NAT, HDIM, NDIM, 0, 1);

    // coord MLP: w = silu(msg @ cW1 + cb1) @ cW2 + cb2
    gemm(msg, cW1pl, cb1l, B1, nullptr, EDG, HDIM, HDIM / 2, 1, 0);      // w1 = B1
    pg_wdot<<<cdiv(EDG, 256), 256, 0, stream>>>(B1, cW2 + l * (HDIM / 2), cb2 + l, wv);

    // coord update
    pg_zero<<<cdiv((long)NAT * 3, 256), 256, 0, stream>>>(delta, NAT * 3);
    pg_coord<<<cdiv(EDG, 256), 256, 0, stream>>>(coords, nbr, wv, mask, delta);
    pg_apply<<<cdiv((long)NAT * 3, 256), 256, 0, stream>>>(coords, delta);
  }

  pg_out<<<cdiv((long)NAT * NDIM + NAT * 3, 256), 256, 0, stream>>>(nf, coords, (float*)d_out);
}
